// PatchAttacker_9414568313316
// MI455X (gfx1250) — compile-verified, matrix operand order checked by reading, not run on hardware
//
#include <hip/hip_runtime.h>
#include <math.h>

// ---------------- problem constants (fixed by setup_inputs) ----------------
#define B_ 16
#define H_ 768
#define W_ 768
#define N_ 16
#define P_ 640
#define DEG20_ 0.34906585039886590f   // 20 deg in radians
#define SQRT2_ 1.41421356237309515f

// ---------------- workspace layout (floats). total ~8.3K floats (~33KB) ----
#define RED_BLKS 64
#define OFF_WMUL  0
#define OFF_BADD  (OFF_WMUL + B_*3)
#define OFF_IMGP  (OFF_BADD + B_*3)
#define OFF_PATP  (OFF_IMGP + B_*RED_BLKS)
#define OFF_SHIFT (OFF_PATP + B_*RED_BLKS)
#define PAR_STRIDE 24
#define OFF_BOX   (OFF_SHIFT + B_)

typedef __attribute__((ext_vector_type(2)))  float    v2f;
typedef __attribute__((ext_vector_type(8)))  float    v8f;
typedef __attribute__((ext_vector_type(16))) _Float16 v16h;

// ---------------- Threefry-2x32 (20 rounds) --------------------------------
__device__ __forceinline__ void tf2x32(unsigned k0, unsigned k1,
                                       unsigned c0, unsigned c1,
                                       unsigned &o0, unsigned &o1) {
  unsigned ks2 = 0x1BD11BDAu ^ k0 ^ k1;
  unsigned x0 = c0 + k0, x1 = c1 + k1;
#define TF_R(r) { x0 += x1; x1 = (x1 << (r)) | (x1 >> (32 - (r))); x1 ^= x0; }
  TF_R(13) TF_R(15) TF_R(26) TF_R(6)   x0 += k1;  x1 += ks2 + 1u;
  TF_R(17) TF_R(29) TF_R(16) TF_R(24)  x0 += ks2; x1 += k0 + 2u;
  TF_R(13) TF_R(15) TF_R(26) TF_R(6)   x0 += k0;  x1 += k1 + 3u;
  TF_R(17) TF_R(29) TF_R(16) TF_R(24)  x0 += k1;  x1 += ks2 + 4u;
  TF_R(13) TF_R(15) TF_R(26) TF_R(6)   x0 += ks2; x1 += k0 + 5u;
#undef TF_R
  o0 = x0; o1 = x1;
}
__device__ __forceinline__ float u01(unsigned r) {
  return __uint_as_float(0x3F800000u | (r >> 9)) - 1.0f;   // [0,1)
}
__device__ __forceinline__ float clipf(float x) {
  return fminf(fmaxf(x, -1.0f), 1.0f);
}

// ---------------- WMMA accumulate: sum of a float4 across the wave ---------
// A = all-ones => D[m][n] = sum_k B[k][n]; all 16 rows identical, so the
// wave-total = (sum over all lanes of all 8 acc elements) / 16.
__device__ __forceinline__ v8f wmma_acc_f4(v8f acc, float4 d) {
#if __has_builtin(__builtin_amdgcn_wmma_f32_16x16x4_f32)
  v2f ones = {1.0f, 1.0f};
  v2f lo = {d.x, d.y};
  v2f hi = {d.z, d.w};
  acc = __builtin_amdgcn_wmma_f32_16x16x4_f32(false, ones, false, lo,
                                              (short)0, acc, false, false);
  acc = __builtin_amdgcn_wmma_f32_16x16x4_f32(false, ones, false, hi,
                                              (short)0, acc, false, false);
#else
  // fallback: codegen-confirmed f16 WMMA, zero-padded B (still exact sums of
  // f16-rounded inputs into f32 accumulators)
  v16h ones; v16h hv;
  #pragma unroll
  for (int i = 0; i < 16; ++i) { ones[i] = (_Float16)1.0f; hv[i] = (_Float16)0.0f; }
  hv[0] = (_Float16)d.x; hv[1] = (_Float16)d.y;
  hv[2] = (_Float16)d.z; hv[3] = (_Float16)d.w;
  acc = __builtin_amdgcn_wmma_f32_16x16x32_f16(false, ones, false, hv,
                                               (short)0, acc, false, false);
#endif
  return acc;
}

__device__ __forceinline__ float block_sum(float s, float* lds8) {
  #pragma unroll
  for (int m = 16; m >= 1; m >>= 1) s += __shfl_xor(s, m, 32);
  int wave = threadIdx.x >> 5, lane = threadIdx.x & 31;
  if (lane == 0) lds8[wave] = s;
  __syncthreads();
  float r = 0.0f;
  if (threadIdx.x == 0)
    for (int w = 0; w < 8; ++w) r += lds8[w];
  return r;   // valid in thread 0 only
}

// ---------------- kernel 0: per-image random gain/bias ---------------------
__global__ void pa_init_params(float* __restrict__ ws) {
  int t = threadIdx.x;
  if (t < B_ * 3) {
    int b = t / 3, c = t % 3;
    unsigned r0, r1;
    tf2x32(0xC0FFEE11u ^ (unsigned)b, 0x9E3779B9u, 0x1000u + (unsigned)c, 0u, r0, r1);
    float uu1 = fmaxf(u01(r0), 1e-7f);
    float uu2 = u01(r1);
    float rad = sqrtf(-2.0f * logf(uu1));
    float z0 = rad * cosf(6.28318530718f * uu2);
    float z1 = rad * sinf(6.28318530718f * uu2);
    ws[OFF_WMUL + t] = 0.5f  + 0.1f  * z0;
    ws[OFF_BADD + t] = 0.01f * z1;
  }
}

// ---------------- kernel 1: image sums via WMMA ----------------------------
__global__ __launch_bounds__(256) void pa_reduce_image(
    const float* __restrict__ images, float* __restrict__ ws) {
  const int b   = blockIdx.x / RED_BLKS;
  const int blk = blockIdx.x % RED_BLKS;
  const long per4 = (long)H_ * W_ * 3 / 4;              // 442368 float4
  const long c4   = per4 / RED_BLKS;                    // 6912 (exact)
  const float4* src = (const float4*)(images + (long)b * H_ * W_ * 3);
  v8f acc = {};
  const int iters = (int)(c4 / 256);                    // 27, uniform trip count
  for (int it = 0; it < iters; ++it) {
    long i = blk * c4 + (long)it * 256 + threadIdx.x;
    acc = wmma_acc_f4(acc, src[i]);
  }
  float s = acc[0]+acc[1]+acc[2]+acc[3]+acc[4]+acc[5]+acc[6]+acc[7];
  __shared__ float lds8[8];
  float tot = block_sum(s, lds8);
  if (threadIdx.x == 0)
    ws[OFF_IMGP + b * RED_BLKS + blk] = tot * (1.0f / 16.0f);
}

// ---------------- kernel 2: adjusted-patch sums via WMMA -------------------
__global__ __launch_bounds__(256) void pa_reduce_patch(
    const float* __restrict__ patch, float* __restrict__ ws) {
  const int b   = blockIdx.x / RED_BLKS;
  const int blk = blockIdx.x % RED_BLKS;
  float w[3], bd[3];
  #pragma unroll
  for (int c = 0; c < 3; ++c) {
    w[c]  = ws[OFF_WMUL + b*3 + c];
    bd[c] = ws[OFF_BADD + b*3 + c];
  }
  const long per4 = (long)P_ * P_ * 3 / 4;              // 307200 float4
  const long c4   = per4 / RED_BLKS;                    // 4800 (exact)
  const float4* src = (const float4*)patch;
  v8f acc = {};
  const int iters = (int)((c4 + 255) / 256);            // 19, uniform (EXEC all-1s)
  for (int it = 0; it < iters; ++it) {
    long i = blk * c4 + (long)it * 256 + threadIdx.x;
    bool ok = (i < (blk + 1) * c4);
    float4 d = ok ? src[i] : make_float4(0.f, 0.f, 0.f, 0.f);
    if (ok) {
      long e = i * 4;
      int c0 = (int)(e % 3);
      int c1 = (c0 + 1) % 3, c2 = (c0 + 2) % 3;
      d.x = clipf(w[c0]*d.x + bd[c0]);
      d.y = clipf(w[c1]*d.y + bd[c1]);
      d.z = clipf(w[c2]*d.z + bd[c2]);
      d.w = clipf(w[c0]*d.w + bd[c0]);
    }
    acc = wmma_acc_f4(acc, d);
  }
  float s = acc[0]+acc[1]+acc[2]+acc[3]+acc[4]+acc[5]+acc[6]+acc[7];
  __shared__ float lds8[8];
  float tot = block_sum(s, lds8);
  if (threadIdx.x == 0)
    ws[OFF_PATP + b * RED_BLKS + blk] = tot * (1.0f / 16.0f);
}

// ---------------- kernel 3: means + per-box geometry -----------------------
__global__ void pa_finalize(const float* __restrict__ boxes,
                            const float* __restrict__ scale_p,
                            float* __restrict__ ws) {
  int t = threadIdx.x;
  if (t < B_) {
    float si = 0.f, sp = 0.f;
    for (int k = 0; k < RED_BLKS; ++k) {
      si += ws[OFF_IMGP + t * RED_BLKS + k];
      sp += ws[OFF_PATP + t * RED_BLKS + k];
    }
    ws[OFF_SHIFT + t] = si * (1.0f / ((float)H_ * W_ * 3))
                      - sp * (1.0f / ((float)P_ * P_ * 3));
  }
  __syncthreads();
  int b = t / N_, n = t % N_;
  const float* bx = boxes + (size_t)(b * N_ + n) * 4;
  float ymin = bx[0], xmin = bx[1], ymax = bx[2], xmax = bx[3];
  float scale = scale_p[0];
  float bh = ymax - ymin, bw = xmax - xmin;
  float longer = fmaxf(bh, bw);
  float ph   = floorf(longer * scale);                  // psize
  float diag = fminf(SQRT2_ * ph, (float)W_);
  unsigned r0, r1, r2, r3, r4, r5;
  unsigned kk0 = 0x9E3779B9u ^ (unsigned)b, kk1 = 0x85EBCA6Bu ^ (unsigned)n;
  tf2x32(kk0, kk1, 1u, 0u, r0, r1);
  tf2x32(kk0, kk1, 2u, 0u, r2, r3);
  tf2x32(kk0, kk1, 3u, 0u, r4, r5);
  const float tol = 0.2f;
  float oy = ymin + bh*0.5f + (u01(r0)*2.f - 1.f) * tol * bh * 0.5f;
  float ox = xmin + bw*0.5f + (u01(r1)*2.f - 1.f) * tol * bw * 0.5f;
  float y0 = fmaxf(oy - diag*0.5f, 0.0f);
  float x0 = fmaxf(ox - diag*0.5f, 0.0f);
  y0 = (y0 + diag > (float)H_) ? (float)H_ - diag : y0;
  x0 = (x0 + diag > (float)W_) ? (float)W_ - diag : x0;
  y0 = floorf(y0); x0 = floorf(x0);
  float d = floorf(diag);
  float valid = (ph * ph > 4.0f) ? 1.0f : 0.0f;
  float top = floorf((d - ph) * 0.5f);
  float angle = (u01(r2)*2.f - 1.f) * DEG20_;
  float delta = (u01(r4)*2.f - 1.f) * 0.3f;
  float* pp = ws + OFF_BOX + (size_t)(b * N_ + n) * PAR_STRIDE;
  pp[0] = y0;  pp[1] = x0;  pp[2] = d;    pp[3] = ph;  pp[4] = top;
  pp[5] = cosf(angle); pp[6] = sinf(angle); pp[7] = (d - 1.0f) * 0.5f;
  pp[8] = (float)P_ / ph;  pp[9] = valid;  pp[10] = delta;
  pp[11] = ws[OFF_SHIFT + b];
  #pragma unroll
  for (int c = 0; c < 3; ++c) {
    pp[12 + c] = ws[OFF_WMUL + b*3 + c];
    pp[15 + c] = ws[OFF_BADD + b*3 + c];
  }
  pp[18] = pp[19] = pp[20] = pp[21] = pp[22] = pp[23] = 0.0f;
}

// ---------------- main kernel: per-pixel compose ---------------------------
__device__ __forceinline__ float sample_adj(const float* __restrict__ patch,
                                            int yi, int xi, int c,
                                            const float* w, const float* bd,
                                            float shift) {
  float t  = patch[((long)yi * P_ + xi) * 3 + c];
  float pv = clipf(w[c] * t + bd[c]);
  return clipf(pv + shift);
}

__global__ __launch_bounds__(256) void pa_main(
    const float* __restrict__ images, const float* __restrict__ patch,
    const float* __restrict__ ws, float* __restrict__ out) {
  __shared__ float lp[N_ * PAR_STRIDE];                 // 384 f32 = 1.5 KB
  const int b = blockIdx.y;
  const float* gpar = ws + OFF_BOX + (size_t)b * N_ * PAR_STRIDE;

  // CDNA5: async DMA of the per-box parameter table into LDS (ASYNCcnt path).
  for (int i = threadIdx.x; i < N_ * PAR_STRIDE; i += blockDim.x) {
    unsigned lds_addr = (unsigned)(uintptr_t)(&lp[i]); // low 32b = LDS offset
    unsigned voff     = (unsigned)(i * 4);
    asm volatile("global_load_async_to_lds_b32 %0, %1, %2"
                 :: "v"(lds_addr), "v"(voff), "s"(gpar) : "memory");
  }

  // Overlap: this thread's image pixel load is issued while the DMA runs.
  const int pix = blockIdx.x * blockDim.x + threadIdx.x;     // < H*W
  const int y = pix / W_, x = pix % W_;
  const long base = ((long)b * H_ * W_ + pix) * 3;
  float i0 = images[base], i1 = images[base + 1], i2 = images[base + 2];

#if __has_builtin(__builtin_amdgcn_s_wait_asynccnt)
  __builtin_amdgcn_s_wait_asynccnt(0);
#else
  asm volatile("s_wait_asynccnt 0" ::: "memory");
#endif
  __syncthreads();

  // Last box whose rotated patch covers this pixel wins. Pixels inside a
  // window but outside the patch get clip(im) in the reference, which is the
  // identity here (all values already in [-1,1]).
  int winner = -1;
  float py = 0.f, px = 0.f, phv = 1.f, sclv = 0.f, deltav = 0.f, shiftv = 0.f;
  float w[3]  = {0.f, 0.f, 0.f};
  float bd[3] = {0.f, 0.f, 0.f};
  for (int n = N_ - 1; n >= 0; --n) {
    const float* pp = lp + n * PAR_STRIDE;
    float y0 = pp[0], x0 = pp[1], d = pp[2];
    float u = (float)y - y0, v = (float)x - x0;
    if (!(u >= 0.f && u < d && v >= 0.f && v < d)) continue;   // cheap early-out
    if (pp[9] == 0.f) continue;                                // invalid box
    float ph = pp[3], top = pp[4], ca = pp[5], sa = pp[6], cc = pp[7];
    float iu = cc + ca * (u - cc) - sa * (v - cc);
    float iv = cc + sa * (u - cc) + ca * (v - cc);
    float ty = iu - top, tx = iv - top;
    if (ty >= 0.f && ty <= ph - 1.f && tx >= 0.f && tx <= ph - 1.f) {
      winner = n; py = ty; px = tx; phv = ph;
      sclv = pp[8]; deltav = pp[10]; shiftv = pp[11];
      w[0] = pp[12]; w[1] = pp[13]; w[2] = pp[14];
      bd[0] = pp[15]; bd[1] = pp[16]; bd[2] = pp[17];
      break;
    }
  }

  float o0 = i0, o1 = i1, o2 = i2;
  if (winner >= 0) {
    (void)phv;
    float sy = (py + 0.5f) * sclv - 0.5f;
    float sx = (px + 0.5f) * sclv - 0.5f;
    float fy = floorf(sy), fx = floorf(sx);
    float wy = sy - fy, wx = sx - fx;
    int y0i = min(max((int)fy,     0), P_ - 1);
    int y1i = min(max((int)fy + 1, 0), P_ - 1);
    int x0i = min(max((int)fx,     0), P_ - 1);
    int x1i = min(max((int)fx + 1, 0), P_ - 1);
    float w00 = (1.f - wy) * (1.f - wx), w01 = (1.f - wy) * wx;
    float w10 = wy * (1.f - wx),         w11 = wy * wx;
    unsigned kk0 = 0xA5A5A5A5u ^ (unsigned)(b * N_ + winner);
    unsigned r0, r1, r2, r3;
    tf2x32(kk0, 0x3C6EF372u, (unsigned)pix, 0u, r0, r1);
    tf2x32(kk0, 0x3C6EF372u, (unsigned)pix, 1u, r2, r3);
    float nz[3] = {(u01(r0)*2.f - 1.f) * 0.01f,
                   (u01(r1)*2.f - 1.f) * 0.01f,
                   (u01(r2)*2.f - 1.f) * 0.01f};
    float res[3];
    #pragma unroll
    for (int c = 0; c < 3; ++c) {
      float v00 = sample_adj(patch, y0i, x0i, c, w, bd, shiftv);
      float v01 = sample_adj(patch, y0i, x1i, c, w, bd, shiftv);
      float v10 = sample_adj(patch, y1i, x0i, c, w, bd, shiftv);
      float v11 = sample_adj(patch, y1i, x1i, c, w, bd, shiftv);
      res[c] = clipf(v00*w00 + v01*w01 + v10*w10 + v11*w11 + nz[c] + deltav);
    }
    o0 = res[0]; o1 = res[1]; o2 = res[2];
  }
  out[base] = o0; out[base + 1] = o1; out[base + 2] = o2;
}

// ---------------- host-side launcher ---------------------------------------
extern "C" void kernel_launch(void* const* d_in, const int* in_sizes, int n_in,
                              void* d_out, int out_size, void* d_ws, size_t ws_size,
                              hipStream_t stream) {
  (void)in_sizes; (void)n_in; (void)out_size; (void)ws_size;
  const float* images = (const float*)d_in[0];   // [B,H,W,3]
  const float* boxes  = (const float*)d_in[1];   // [B,N,4]
  const float* patch  = (const float*)d_in[2];   // [P,P,3]
  const float* scale  = (const float*)d_in[3];   // [1]
  float* out = (float*)d_out;
  float* ws  = (float*)d_ws;                     // needs ~33 KB

  hipLaunchKernelGGL(pa_init_params, dim3(1), dim3(256), 0, stream, ws);
  hipLaunchKernelGGL(pa_reduce_image, dim3(B_ * RED_BLKS), dim3(256), 0, stream,
                     images, ws);
  hipLaunchKernelGGL(pa_reduce_patch, dim3(B_ * RED_BLKS), dim3(256), 0, stream,
                     patch, ws);
  hipLaunchKernelGGL(pa_finalize, dim3(1), dim3(256), 0, stream, boxes, scale, ws);
  hipLaunchKernelGGL(pa_main, dim3((H_ * W_) / 256, B_), dim3(256), 0, stream,
                     images, patch, ws, out);
}
